// MultiheadAttention_38122129719574
// MI455X (gfx1250) — matrix operand order repaired and checked by reading, not compiled, not run on hardware
//
#include <hip/hip_runtime.h>
#include <hip/hip_bf16.h>

typedef __bf16 bf16;
typedef __bf16 v16bf __attribute__((ext_vector_type(16)));
typedef __bf16 v8bf  __attribute__((ext_vector_type(8)));
typedef __bf16 v4bf  __attribute__((ext_vector_type(4)));
typedef float  v8f   __attribute__((ext_vector_type(8)));

constexpr int Bc = 2, Lc = 2048, Ec = 1024, Hc = 16, Dc = 64;
constexpr int Mc = Bc * Lc;   // 4096 rows in flattened [B*L, E]

// ---------------------------------------------------------------------------
// Pack X fp32 -> bf16 (same [M, E] row-major layout). 4 elements per thread.
// ---------------------------------------------------------------------------
__global__ __launch_bounds__(256)
void pack_x_kernel(const float* __restrict__ X, bf16* __restrict__ Xb) {
  const size_t i = ((size_t)blockIdx.x * blockDim.x + threadIdx.x) * 4;
  const float4 v = *(const float4*)(X + i);
  v4bf o;
  o[0] = (bf16)v.x; o[1] = (bf16)v.y; o[2] = (bf16)v.z; o[3] = (bf16)v.w;
  *(v4bf*)(Xb + i) = o;
}

// ---------------------------------------------------------------------------
// Transpose + convert weight: W fp32 [K=1024, N=1024] -> Wt bf16 [N, K].
// Classic 32x32 LDS tile transpose, block (32, 8).
// ---------------------------------------------------------------------------
__global__ __launch_bounds__(256)
void pack_wt_kernel(const float* __restrict__ W, bf16* __restrict__ Wt) {
  __shared__ float t[32][33];
  int x = blockIdx.x * 32 + threadIdx.x;      // n (input col)
  int y = blockIdx.y * 32 + threadIdx.y;      // k (input row)
#pragma unroll
  for (int i = 0; i < 32; i += 8)
    t[threadIdx.y + i][threadIdx.x] = W[(size_t)(y + i) * Ec + x];
  __syncthreads();
  x = blockIdx.y * 32 + threadIdx.x;          // k (output col)
  y = blockIdx.x * 32 + threadIdx.y;          // n (output row)
#pragma unroll
  for (int i = 0; i < 32; i += 8)
    Wt[(size_t)(y + i) * Ec + x] = (bf16)t[threadIdx.x][threadIdx.y + i];
}

// ---------------------------------------------------------------------------
// Fused projection GEMM + (optional) interleaved RoPE + layout-transform store.
// Y = Xb[4096,1024](bf16) * Wt^T (Wt is bf16 [N,K]), f32 accumulate via WMMA.
// mode 0: Q  -> rope, store bf16 [B,H,L,D]
// mode 1: K  -> rope, store bf16 [B,H,L,D]
// mode 2: V  -> no rope, store bf16 transposed [B,H,D,L]
// Block: 128 threads = 4 waves. Wave w computes rows [tile+16w, +16) x 64 cols.
// ---------------------------------------------------------------------------
__global__ __launch_bounds__(128)
void proj_rope_kernel(const bf16* __restrict__ Xb, const bf16* __restrict__ Wt,
                      bf16* __restrict__ out, int mode) {
  const int lane = threadIdx.x & 31;
  const int wave = threadIdx.x >> 5;
  const int hi   = lane >> 4;       // 0: lanes 0-15, 1: lanes 16-31
  const int ln   = lane & 15;
  const int m0   = blockIdx.x * 64 + wave * 16;   // 16 output rows for this wave
  const int n0   = blockIdx.y * 64;               // 64 output cols

  v8f acc[4] = {};

  for (int kk = 0; kk < Ec; kk += 32) {
    // A fragment (16x32 bf16): row = m0+ln; K contiguous {hi*16..+15}
    const bf16* arow = Xb + (size_t)(m0 + ln) * Ec + kk + hi * 16;
    v8bf alo = *(const v8bf*)arow;
    v8bf ahi = *(const v8bf*)(arow + 8);
    v16bf a;
#pragma unroll
    for (int i = 0; i < 8; ++i) { a[i] = alo[i]; a[i + 8] = ahi[i]; }
#pragma unroll
    for (int nb = 0; nb < 4; ++nb) {
      // B fragment (32x16): col n = n0+nb*16+ln -> Wt row n, contiguous K
      const bf16* brow = Wt + (size_t)(n0 + nb * 16 + ln) * Ec + kk + hi * 16;
      v16bf bfr = *(const v16bf*)brow;
      acc[nb] = __builtin_amdgcn_wmma_f32_16x16x32_bf16(
          false, a, false, bfr, (short)0, acc[nb], false, false);
    }
  }

  // Epilogue: interleaved RoPE (modes 0,1) + store. C layout: lane ln = col,
  // VGPR idx -> local row idx + 8*hi.
#pragma unroll
  for (int nb = 0; nb < 4; ++nb) {
    const int f = n0 + nb * 16 + ln;      // feature index in [0,1024)
    const int h = f >> 6;
    const int d = f & 63;
    const float theta = __powf(10000.0f, -(float)(d & ~1) / 64.0f);
#pragma unroll
    for (int idx = 0; idx < 8; ++idx) {
      const int mrow = m0 + idx + 8 * hi;
      const int b = mrow >> 11;           // / 2048
      const int l = mrow & 2047;
      float val = acc[nb][idx];
      if (mode != 2) {
        // partner column (d^1) lives in lane ln^1 (same row set)
        float partner = __shfl_xor(val, 1);
        float rot = (d & 1) ? partner : -partner;   // t2 = [-t1,t0,-t3,t2,...]
        float ang = (float)l * theta;
        val = val * __cosf(ang) + rot * __sinf(ang);
      }
      const bf16 bv = (bf16)val;
      if (mode == 2) {
        out[(((size_t)b * Hc + h) * Dc + d) * Lc + l] = bv;   // Vt [B,H,D,L]
      } else {
        out[(((size_t)b * Hc + h) * Lc + l) * Dc + d] = bv;   // [B,H,L,D]
      }
    }
  }
}

// ---------------------------------------------------------------------------
// Flash attention: one (b,h,64-query tile) per workgroup; 4 waves x 16 rows.
// S = Q K^T via WMMA -> LDS -> online softmax (half-row per lane) -> P bf16
// in LDS -> O += P * V^T via WMMA. Causal mask applied analytically.
// Ctx written bf16 [B, L, E] for the output projection.
// ---------------------------------------------------------------------------
__global__ __launch_bounds__(128)
void attn_kernel(const bf16* __restrict__ Q, const bf16* __restrict__ K,
                 const bf16* __restrict__ Vt, bf16* __restrict__ Ctx) {
  const int lane = threadIdx.x & 31;
  const int wave = threadIdx.x >> 5;
  const int hi   = lane >> 4;
  const int ln   = lane & 15;
  const int bh   = blockIdx.y;            // b*H + h
  const int b    = bh >> 4;
  const int h    = bh & 15;
  const int q0   = blockIdx.x * 64;
  const int qr0  = q0 + wave * 16;        // this wave's 16 query rows

  __shared__ __align__(16) float Sbuf[4][16 * 64];
  __shared__ __align__(16) bf16  Pbuf[4][16 * 64];
  __shared__ float Arow[4][16];
  __shared__ float Erow[4][16];

  const bf16* Qb = Q  + (size_t)bh * Lc * Dc;
  const bf16* Kb = K  + (size_t)bh * Lc * Dc;
  const bf16* Vb = Vt + (size_t)bh * Dc * Lc;

  // Q A-fragments for this wave's 16 rows (2 fragments cover D=64).
  v16bf qa[2];
#pragma unroll
  for (int c = 0; c < 2; ++c) {
    const bf16* p = Qb + (size_t)(qr0 + ln) * Dc + c * 32 + hi * 16;
    v8bf lo = *(const v8bf*)p;
    v8bf hh = *(const v8bf*)(p + 8);
#pragma unroll
    for (int i = 0; i < 8; ++i) { qa[c][i] = lo[i]; qa[c][i + 8] = hh[i]; }
  }

  v8f O[4] = {};
  float mrun = -3.0e38f;
  float ell  = 0.0f;
  const float scale = 0.125f;             // 1/sqrt(64)
  const int jmax = (q0 + 63) >> 6;        // inclusive causal tile bound

  for (int j = 0; j <= jmax; ++j) {
    const int k0 = j * 64;
    if (j < jmax)  // hint next K tile toward the caches (global_prefetch_b8)
      __builtin_prefetch(Kb + (size_t)(k0 + 64 + lane * 2) * Dc, 0, 1);

    // ---- S = Q * K^T (16 q-rows x 64 keys) ----
    v8f S[4] = {};
#pragma unroll
    for (int nb = 0; nb < 4; ++nb) {
#pragma unroll
      for (int c = 0; c < 2; ++c) {
        const bf16* p = Kb + (size_t)(k0 + nb * 16 + ln) * Dc + c * 32 + hi * 16;
        v16bf bfr = *(const v16bf*)p;
        S[nb] = __builtin_amdgcn_wmma_f32_16x16x32_bf16(
            false, qa[c], false, bfr, (short)0, S[nb], false, false);
      }
    }
    // scale + causal mask + transpose through LDS
#pragma unroll
    for (int nb = 0; nb < 4; ++nb) {
      const int kcol = k0 + nb * 16 + ln;
#pragma unroll
      for (int idx = 0; idx < 8; ++idx) {
        const int qrow = qr0 + idx + 8 * hi;
        float sv = S[nb][idx] * scale;
        if (kcol > qrow) sv = -3.0e38f;
        Sbuf[wave][(idx + 8 * hi) * 64 + nb * 16 + ln] = sv;
      }
    }
    __syncthreads();

    // ---- online softmax: lane handles row ln, cols [hi*32, +32) ----
    {
      const float* srow = &Sbuf[wave][ln * 64 + hi * 32];
      float tmax = -3.0e38f;
#pragma unroll
      for (int i = 0; i < 32; ++i) tmax = fmaxf(tmax, srow[i]);
      tmax = fmaxf(tmax, __shfl_xor(tmax, 16));
      const float newm  = fmaxf(mrun, tmax);
      const float alpha = __expf(mrun - newm);
      float psum = 0.0f;
      bf16* prow = &Pbuf[wave][ln * 64 + hi * 32];
#pragma unroll
      for (int i = 0; i < 32; ++i) {
        float p = __expf(srow[i] - newm);
        psum += p;
        prow[i] = (bf16)p;
      }
      psum += __shfl_xor(psum, 16);
      ell  = alpha * ell + psum;
      mrun = newm;
      Arow[wave][ln] = alpha;   // lanes ln and ln+16 write identical value
    }
    __syncthreads();

    // ---- rescale O, then O += P * V ----
    float al[8];
#pragma unroll
    for (int idx = 0; idx < 8; ++idx) al[idx] = Arow[wave][idx + 8 * hi];
#pragma unroll
    for (int nb = 0; nb < 4; ++nb)
#pragma unroll
      for (int idx = 0; idx < 8; ++idx) O[nb][idx] *= al[idx];

    v16bf pa[2];
#pragma unroll
    for (int c = 0; c < 2; ++c) {
      const bf16* p = &Pbuf[wave][ln * 64 + c * 32 + hi * 16];
      v8bf lo = *(const v8bf*)p;
      v8bf hh = *(const v8bf*)(p + 8);
#pragma unroll
      for (int i = 0; i < 8; ++i) { pa[c][i] = lo[i]; pa[c][i + 8] = hh[i]; }
    }
#pragma unroll
    for (int nb = 0; nb < 4; ++nb) {
#pragma unroll
      for (int c = 0; c < 2; ++c) {
        // B frag: col n = d (nb*16+ln), K = keys, contiguous in Vt rows
        const bf16* p = Vb + (size_t)(nb * 16 + ln) * Lc + k0 + c * 32 + hi * 16;
        v16bf bfr = *(const v16bf*)p;
        O[nb] = __builtin_amdgcn_wmma_f32_16x16x32_bf16(
            false, pa[c], false, bfr, (short)0, O[nb], false, false);
      }
    }
    __syncthreads();
  }

  // ---- normalize by row sums and write Ctx [B, L, H*64+d] as bf16 ----
  Erow[wave][ln] = ell;
  __syncthreads();
  float el[8];
#pragma unroll
  for (int idx = 0; idx < 8; ++idx) el[idx] = 1.0f / Erow[wave][idx + 8 * hi];
#pragma unroll
  for (int nb = 0; nb < 4; ++nb) {
    const int d = nb * 16 + ln;
#pragma unroll
    for (int idx = 0; idx < 8; ++idx) {
      const int qrow = qr0 + idx + 8 * hi;
      Ctx[((size_t)b * Lc + qrow) * Ec + h * Dc + d] = (bf16)(O[nb][idx] * el[idx]);
    }
  }
}

// ---------------------------------------------------------------------------
// Output projection: Y[4096,1024] f32 = Ctx(bf16) * WoT^T + bo
// WoT is pre-transposed bf16 [N, K].
// ---------------------------------------------------------------------------
__global__ __launch_bounds__(128)
void outproj_kernel(const bf16* __restrict__ Ctx, const bf16* __restrict__ WoT,
                    const float* __restrict__ bo, float* __restrict__ Y) {
  const int lane = threadIdx.x & 31;
  const int wave = threadIdx.x >> 5;
  const int hi   = lane >> 4;
  const int ln   = lane & 15;
  const int m0   = blockIdx.x * 64 + wave * 16;
  const int n0   = blockIdx.y * 64;

  v8f acc[4] = {};
  for (int kk = 0; kk < Ec; kk += 32) {
    const bf16* arow = Ctx + (size_t)(m0 + ln) * Ec + kk + hi * 16;
    v8bf alo = *(const v8bf*)arow;
    v8bf ahi = *(const v8bf*)(arow + 8);
    v16bf a;
#pragma unroll
    for (int i = 0; i < 8; ++i) { a[i] = alo[i]; a[i + 8] = ahi[i]; }
#pragma unroll
    for (int nb = 0; nb < 4; ++nb) {
      const bf16* brow = WoT + (size_t)(n0 + nb * 16 + ln) * Ec + kk + hi * 16;
      v16bf bfr = *(const v16bf*)brow;
      acc[nb] = __builtin_amdgcn_wmma_f32_16x16x32_bf16(
          false, a, false, bfr, (short)0, acc[nb], false, false);
    }
  }
#pragma unroll
  for (int nb = 0; nb < 4; ++nb) {
    const int f = n0 + nb * 16 + ln;
    const float bias = bo[f];
#pragma unroll
    for (int idx = 0; idx < 8; ++idx) {
      const int mrow = m0 + idx + 8 * hi;
      Y[(size_t)mrow * Ec + f] = acc[nb][idx] + bias;
    }
  }
}

// ---------------------------------------------------------------------------
extern "C" void kernel_launch(void* const* d_in, const int* in_sizes, int n_in,
                              void* d_out, int out_size, void* d_ws, size_t ws_size,
                              hipStream_t stream) {
  (void)in_sizes; (void)n_in; (void)out_size; (void)ws_size;
  const float* x  = (const float*)d_in[0];
  // d_in[1] = causal mask (bool) — applied analytically in attn_kernel
  const float* Wq = (const float*)d_in[2];
  const float* Wk = (const float*)d_in[3];
  const float* Wv = (const float*)d_in[4];
  const float* Wo = (const float*)d_in[5];
  const float* bo = (const float*)d_in[6];
  float* out = (float*)d_out;

  char* ws = (char*)d_ws;
  const size_t XSZ = (size_t)Mc * Ec * sizeof(bf16);           // 8 MB
  const size_t WSZ = (size_t)Ec * Ec * sizeof(bf16);           // 2 MB
  const size_t QSZ = (size_t)Bc * Hc * Lc * Dc * sizeof(bf16); // 8 MB
  bf16* Xb  = (bf16*)(ws);
  bf16* WqT = (bf16*)(ws + XSZ);
  bf16* WkT = (bf16*)(ws + XSZ + WSZ);
  bf16* WvT = (bf16*)(ws + XSZ + 2 * WSZ);
  bf16* WoT = (bf16*)(ws + XSZ + 3 * WSZ);
  bf16* Qb  = (bf16*)(ws + XSZ + 4 * WSZ);
  bf16* Kb  = (bf16*)(ws + XSZ + 4 * WSZ + QSZ);
  bf16* Vtb = (bf16*)(ws + XSZ + 4 * WSZ + 2 * QSZ);
  bf16* Ctx = (bf16*)(ws + XSZ + 4 * WSZ + 3 * QSZ);

  // Stage 0: pack inputs to bf16 (weights transposed for contiguous B-frags)
  pack_x_kernel<<<(Mc * Ec) / (256 * 4), 256, 0, stream>>>(x, Xb);
  dim3 gT(Ec / 32, Ec / 32);
  dim3 bT(32, 8);
  pack_wt_kernel<<<gT, bT, 0, stream>>>(Wq, WqT);
  pack_wt_kernel<<<gT, bT, 0, stream>>>(Wk, WkT);
  pack_wt_kernel<<<gT, bT, 0, stream>>>(Wv, WvT);
  pack_wt_kernel<<<gT, bT, 0, stream>>>(Wo, WoT);

  // Stage 1: projections (+RoPE for Q,K; V stored transposed)
  dim3 gGemm(Mc / 64, Ec / 64);   // 64 x 16 workgroups
  proj_rope_kernel<<<gGemm, 128, 0, stream>>>(Xb, WqT, Qb, 0);
  proj_rope_kernel<<<gGemm, 128, 0, stream>>>(Xb, WkT, Kb, 1);
  proj_rope_kernel<<<gGemm, 128, 0, stream>>>(Xb, WvT, Vtb, 2);

  // Stage 2: fused causal flash attention
  dim3 gAttn(Lc / 64, Bc * Hc);   // 32 x 32 workgroups
  attn_kernel<<<gAttn, 128, 0, stream>>>(Qb, Kb, Vtb, Ctx);

  // Stage 3: output projection + bias
  outproj_kernel<<<gGemm, 128, 0, stream>>>(Ctx, WoT, bo, out);
}
